// Tacotron2Decoder_21560735826323
// MI455X (gfx1250) — matrix operand order repaired and checked by reading, not compiled, not run on hardware
//
#include <hip/hip_runtime.h>

typedef _Float16 h16;
typedef __attribute__((ext_vector_type(16))) _Float16 v16h;
typedef __attribute__((ext_vector_type(8)))  float v8f;
typedef __attribute__((ext_vector_type(4)))  unsigned int u32x4;
typedef int i32x4v __attribute__((vector_size(16)));  // matches builtin param type

// gfx1250 async global->LDS copy (ASYNCcnt-tracked), with safe fallback.
#if defined(__has_builtin)
# if __has_builtin(__builtin_amdgcn_global_load_async_to_lds_b128)
#  define HAVE_ASYNC_COPY 1
# endif
# if __has_builtin(__builtin_amdgcn_s_wait_asynccnt)
#  define HAVE_WAIT_ASYNC 1
# endif
#endif
#ifndef HAVE_ASYNC_COPY
# define HAVE_ASYNC_COPY 0
#endif
#ifndef HAVE_WAIT_ASYNC
# define HAVE_WAIT_ASYNC 0
#endif

union FragH {
  v16h h;
  u32x4 u[2];
};
union H8 {
  h16 v[8];
  u32x4 u;
};

#define WMMA_F16(a, b, c) \
  __builtin_amdgcn_wmma_f32_16x16x32_f16(false, (a), false, (b), (short)0, (c), false, false)

static __device__ __forceinline__ float sigmoidf_(float x) {
  return 1.0f / (1.0f + __expf(-x));
}

// ---------------- elementwise helpers ----------------
__global__ void cvt_f32_to_f16(const float* __restrict__ src, h16* __restrict__ dst, int n) {
  int i = blockIdx.x * blockDim.x + threadIdx.x;
  if (i < n) dst[i] = (h16)src[i];
}

// (R x C) f32 -> (ceil-padded) f16 with zero fill; dst is (n/Cp) x Cp
__global__ void cvt_pad_f16(const float* __restrict__ src, h16* __restrict__ dst,
                            int R, int C, int Cp, int n) {
  int i = blockIdx.x * blockDim.x + threadIdx.x;
  if (i < n) {
    int r = i / Cp, c = i - r * Cp;
    dst[i] = (r < R && c < C) ? (h16)src[r * C + c] : (h16)0.0f;
  }
}

// prev[b,t,:] = (t==0) ? 0 : mel[b,t-1,:], stored with K padded 80 -> 96
__global__ void build_prev(const float* __restrict__ mel, h16* __restrict__ prev, int n) {
  int i = blockIdx.x * blockDim.x + threadIdx.x;
  if (i < n) {
    int bt = i / 96, m = i - bt * 96;
    int tt = bt & 1023;  // T = 1024
    prev[i] = (m < 80 && tt != 0) ? (h16)mel[(size_t)bt * 80 - 80 + m] : (h16)0.0f;
  }
}

// stage a 128x32 f16 A tile into LDS (pad-40 row stride); 2 x b128 per thread.
// Async path: global_load_async_to_lds_b128 (no VGPR round-trip, ASYNCcnt).
static __device__ __forceinline__ void stage_a(const h16* __restrict__ A, int lda,
                                               int row0, int kb, int t, h16* dst) {
#pragma unroll
  for (int i = 0; i < 2; ++i) {
    int u = t * 2 + i;
    int r = u >> 2, q = u & 3;
    const h16* src = A + (size_t)(row0 + r) * lda + kb + q * 8;
    h16* d = dst + r * 40 + q * 8;
#if HAVE_ASYNC_COPY
    __builtin_amdgcn_global_load_async_to_lds_b128((i32x4v*)src, (i32x4v*)d, 0, 0);
#else
    *(u32x4*)d = *(const u32x4*)src;
    __builtin_prefetch(src + 64, 0, 1);  // global_prefetch_b8 (speculative: safe)
#endif
  }
}

// drain this wave's outstanding async LDS copies before the block barrier
static __device__ __forceinline__ void async_fence() {
#if HAVE_ASYNC_COPY
# if HAVE_WAIT_ASYNC
  __builtin_amdgcn_s_wait_asynccnt(0);
# else
  asm volatile("s_wait_asynccnt 0x0" ::: "memory");
# endif
#endif
}

// ---------------- generic GEMM: C = act(A @ W^T + bias) ----------------
// Preconditions: K % 32 == 0, W padded to >= gridDim.y*128 rows, rows % 128 == 0.
// Block tile 128x128, 8 waves, each wave 32x64; A double-buffered in LDS via
// async copies; all 4 B fragments preloaded per k-step so waits amortize.
__global__ __launch_bounds__(256) void gemm_kernel(
    const h16* __restrict__ A, int lda,
    const h16* __restrict__ W,
    const float* __restrict__ bias,
    int N, int K, int act,
    h16* __restrict__ out16, float* __restrict__ out32, int ldc) {
  __shared__ __align__(16) h16 At[2][128 * 40];
  const int t = threadIdx.x;
  const int lane = t & 31, wave = t >> 5;
  const int wm = wave >> 1, wn = wave & 1;
  const int lh = lane >> 4, lm = lane & 15;
  const int row0 = blockIdx.x * 128;
  const int col0 = blockIdx.y * 128;

  // W row base pointers, hoisted out of the k-loop
  const h16* wrow[4];
#pragma unroll
  for (int ct = 0; ct < 4; ++ct)
    wrow[ct] = W + (size_t)(col0 + wn * 64 + ct * 16 + lm) * K + lh * 8;

  v8f acc[2][4] = {};
  stage_a(A, lda, row0, 0, t, At[0]);

  int buf = 0;
  for (int kb = 0; kb < K; kb += 32, buf ^= 1) {
    async_fence();
    __syncthreads();
    if (kb + 32 < K) stage_a(A, lda, row0, kb + 32, t, At[buf ^ 1]);

    FragH b[4];
#pragma unroll
    for (int ct = 0; ct < 4; ++ct) {
      const h16* wp = wrow[ct] + kb;
      b[ct].u[0] = *(const u32x4*)wp;
      b[ct].u[1] = *(const u32x4*)(wp + 16);
    }
    FragH a[2];
#pragma unroll
    for (int mt = 0; mt < 2; ++mt) {
      const h16* s = At[buf] + (wm * 32 + mt * 16 + lm) * 40;
      a[mt].u[0] = *(const u32x4*)(s + 8 * lh);
      a[mt].u[1] = *(const u32x4*)(s + 16 + 8 * lh);
    }
#pragma unroll
    for (int ct = 0; ct < 4; ++ct)
#pragma unroll
      for (int mt = 0; mt < 2; ++mt)
        acc[mt][ct] = WMMA_F16(a[mt].h, b[ct].h, acc[mt][ct]);
  }

#pragma unroll
  for (int mt = 0; mt < 2; ++mt)
#pragma unroll
    for (int ct = 0; ct < 4; ++ct) {
      int col = col0 + wn * 64 + ct * 16 + lm;
      if (col >= N) continue;
      float bv = bias ? bias[col] : 0.0f;
#pragma unroll
      for (int g = 0; g < 8; ++g) {
        int row = row0 + wm * 32 + mt * 16 + g + 8 * lh;  // C layout: M = g + 8*lanehi
        float v = acc[mt][ct][g] + bv;
        if (act == 1) v = fmaxf(v, 0.0f);
        if (out16) out16[(size_t)row * ldc + col] = (h16)v;
        if (out32) out32[(size_t)row * ldc + col] = v;
      }
    }
}

// ---------------- fused LSTM-cell GEMM (zero initial state) ----------------
// h = sigmoid(o)*tanh(sigmoid(i)*tanh(g)); f gate and W_hh terms are dead
// (h0=c0=0). Computes only i/g/o gate columns; gates never hit memory.
__global__ __launch_bounds__(256) void lstm_kernel(
    const h16* __restrict__ A, int lda,
    const h16* __restrict__ Wih, int K,
    const float* __restrict__ b_ih, const float* __restrict__ b_hh,
    h16* __restrict__ outH) {
  const int H = 512;
  __shared__ __align__(16) h16 At[2][128 * 40];
  const int t = threadIdx.x;
  const int lane = t & 31, wave = t >> 5;
  const int wm = wave >> 1, wn = wave & 1;
  const int lh = lane >> 4, lm = lane & 15;
  const int row0 = blockIdx.x * 128;
  const int col0 = blockIdx.y * 64;
  const int goff[3] = {0, 2 * H, 3 * H};  // i, g, o rows inside w_ih

  const h16* wrow[3][2];
#pragma unroll
  for (int G = 0; G < 3; ++G)
#pragma unroll
    for (int ct = 0; ct < 2; ++ct)
      wrow[G][ct] = Wih +
          (size_t)(goff[G] + col0 + wn * 32 + ct * 16 + lm) * K + lh * 8;

  v8f acc[3][2][2] = {};
  stage_a(A, lda, row0, 0, t, At[0]);

  int buf = 0;
  for (int kb = 0; kb < K; kb += 32, buf ^= 1) {
    async_fence();
    __syncthreads();
    if (kb + 32 < K) stage_a(A, lda, row0, kb + 32, t, At[buf ^ 1]);

    FragH a[2];
#pragma unroll
    for (int mt = 0; mt < 2; ++mt) {
      const h16* s = At[buf] + (wm * 32 + mt * 16 + lm) * 40;
      a[mt].u[0] = *(const u32x4*)(s + 8 * lh);
      a[mt].u[1] = *(const u32x4*)(s + 16 + 8 * lh);
    }
#pragma unroll
    for (int G = 0; G < 3; ++G) {
      FragH b[2];
#pragma unroll
      for (int ct = 0; ct < 2; ++ct) {
        const h16* wp = wrow[G][ct] + kb;
        b[ct].u[0] = *(const u32x4*)wp;
        b[ct].u[1] = *(const u32x4*)(wp + 16);
      }
#pragma unroll
      for (int ct = 0; ct < 2; ++ct)
#pragma unroll
        for (int mt = 0; mt < 2; ++mt)
          acc[G][mt][ct] = WMMA_F16(a[mt].h, b[ct].h, acc[G][mt][ct]);
    }
  }

#pragma unroll
  for (int mt = 0; mt < 2; ++mt)
#pragma unroll
    for (int ct = 0; ct < 2; ++ct) {
      int col = col0 + wn * 32 + ct * 16 + lm;
      float bi = b_ih[col] + b_hh[col];
      float bg = b_ih[2 * H + col] + b_hh[2 * H + col];
      float bo = b_ih[3 * H + col] + b_hh[3 * H + col];
#pragma unroll
      for (int g = 0; g < 8; ++g) {
        int row = row0 + wm * 32 + mt * 16 + g + 8 * lh;
        float iv = acc[0][mt][ct][g] + bi;
        float gv = acc[1][mt][ct][g] + bg;
        float ov = acc[2][mt][ct][g] + bo;
        float c = sigmoidf_(iv) * tanhf(gv);
        float hh = sigmoidf_(ov) * tanhf(c);
        outH[(size_t)row * H + col] = (h16)hh;
      }
    }
}

// ---------------- flash attention ----------------
// One block = one (b,h) pair x 128 query rows (8 waves x 16 rows each).
// QK^T B-frags load K rows directly (two 8-wide K runs per lane, at 8*lh and 16+8*lh);
// V is scatter-transposed into double-buffered LDS for PV; P goes
// C-layout -> per-wave LDS -> A-layout.
__global__ __launch_bounds__(256) void attn_kernel(
    const h16* __restrict__ qkv, h16* __restrict__ ctx) {
  const int T = 1024;
  __shared__ __align__(16) h16 VtT[2][64 * 40];  // [dim][key] transposed V tiles
  __shared__ __align__(16) h16 Pl[8 * 16 * 40];  // per-wave 16x32 P tiles
  const int t = threadIdx.x;
  const int lane = t & 31, wave = t >> 5;
  const int lh = lane >> 4, lm = lane & 15;
  const int b = blockIdx.x >> 3, h = blockIdx.x & 7;
  const int t0 = blockIdx.y * 128;
  const size_t rb = (size_t)b * T;
  const int qcol = h * 64, kcol = 512 + h * 64, vcol = 1024 + h * 64;

  FragH q[2];
  {
    const h16* s = qkv + (rb + t0 + wave * 16 + lm) * 1536 + qcol;
    q[0].u[0] = *(const u32x4*)(s + 8 * lh);
    q[0].u[1] = *(const u32x4*)(s + 16 + 8 * lh);
    q[1].u[0] = *(const u32x4*)(s + 32 + 8 * lh);
    q[1].u[1] = *(const u32x4*)(s + 48 + 8 * lh);
  }

  // K-row base pointers (advance by kb0*1536 in the loop)
  const h16* krow[2][2];
#pragma unroll
  for (int ct = 0; ct < 2; ++ct)
#pragma unroll
    for (int kk = 0; kk < 2; ++kk)
      krow[ct][kk] =
          qkv + (rb + ct * 16 + lm) * 1536 + kcol + kk * 32 + lh * 8;

  v8f accO[4] = {};
  float mrow[8], lrow[8];
#pragma unroll
  for (int g = 0; g < 8; ++g) {
    mrow[g] = -1e30f;
    lrow[g] = 0.0f;
  }

  const int vkey = t & 31, vd8 = (t >> 5) * 8;
  const h16* vbase = qkv + (rb + vkey) * 1536 + vcol + vd8;
  {  // prologue: stage V tile for kb0 = 0
    H8 hv;
    hv.u = *(const u32x4*)vbase;
#pragma unroll
    for (int j = 0; j < 8; ++j) VtT[0][(vd8 + j) * 40 + vkey] = hv.v[j];
  }

  int buf = 0;
  for (int kb0 = 0; kb0 < T; kb0 += 32, buf ^= 1) {
    __syncthreads();
    if (kb0 + 32 < T) {  // stage next transposed V tile
      H8 hv;
      hv.u = *(const u32x4*)(vbase + (size_t)(kb0 + 32) * 1536);
#pragma unroll
      for (int j = 0; j < 8; ++j) VtT[buf ^ 1][(vd8 + j) * 40 + vkey] = hv.v[j];
    }

    // S = Q K^T for 16 queries x 32 keys
    v8f s[2] = {};
#pragma unroll
    for (int ct = 0; ct < 2; ++ct) {
#pragma unroll
      for (int kk = 0; kk < 2; ++kk) {
        FragH bk;
        const h16* kp = krow[ct][kk] + (size_t)kb0 * 1536;
        bk.u[0] = *(const u32x4*)kp;
        bk.u[1] = *(const u32x4*)(kp + 16);
        s[ct] = WMMA_F16(q[kk].h, bk.h, s[ct]);
      }
    }

    // online softmax (scale = 1/sqrt(64) = 0.125), per C-layout row g+8*lh
#pragma unroll
    for (int g = 0; g < 8; ++g) {
      float s0 = s[0][g] * 0.125f;
      float s1 = s[1][g] * 0.125f;
      float vmax = fmaxf(s0, s1);
#pragma unroll
      for (int off = 1; off < 16; off <<= 1)
        vmax = fmaxf(vmax, __shfl_xor(vmax, off, 32));
      float mnew = fmaxf(mrow[g], vmax);
      float corr = __expf(mrow[g] - mnew);
      float p0 = __expf(s0 - mnew);
      float p1 = __expf(s1 - mnew);
      float ps = p0 + p1;
#pragma unroll
      for (int off = 1; off < 16; off <<= 1) ps += __shfl_xor(ps, off, 32);
      lrow[g] = lrow[g] * corr + ps;
      mrow[g] = mnew;
#pragma unroll
      for (int ct = 0; ct < 4; ++ct) accO[ct][g] *= corr;
      int prow = g + 8 * lh;
      Pl[wave * 640 + prow * 40 + lm] = (h16)p0;
      Pl[wave * 640 + prow * 40 + 16 + lm] = (h16)p1;
    }

    // O += P @ V
    FragH p;
    {
      const h16* s2 = Pl + wave * 640 + lm * 40;
      p.u[0] = *(const u32x4*)(s2 + 8 * lh);
      p.u[1] = *(const u32x4*)(s2 + 16 + 8 * lh);
    }
#pragma unroll
    for (int ct = 0; ct < 4; ++ct) {
      FragH bv;
      const h16* vp = VtT[buf] + (ct * 16 + lm) * 40 + lh * 8;
      bv.u[0] = *(const u32x4*)vp;
      bv.u[1] = *(const u32x4*)(vp + 16);
      accO[ct] = WMMA_F16(p.h, bv.h, accO[ct]);
    }
  }

#pragma unroll
  for (int g = 0; g < 8; ++g) {
    float inv = 1.0f / lrow[g];
    int row = t0 + wave * 16 + g + 8 * lh;
#pragma unroll
    for (int ct = 0; ct < 4; ++ct)
      ctx[(rb + row) * 512 + h * 64 + ct * 16 + lm] = (h16)(accO[ct][g] * inv);
  }
}

// ---------------- host orchestration ----------------
extern "C" void kernel_launch(void* const* d_in, const int* in_sizes, int n_in,
                              void* d_out, int out_size, void* d_ws, size_t ws_size,
                              hipStream_t stream) {
  (void)in_sizes; (void)n_in; (void)out_size; (void)ws_size;
  const int B = 16, T = 1024;
  const int R = B * T;  // 16384 rows

  const float* enc_f32    = (const float*)d_in[0];
  const float* mel_f32    = (const float*)d_in[1];
  const float* w_prenet   = (const float*)d_in[2];
  const float* b_prenet   = (const float*)d_in[3];
  const float* in_proj_w  = (const float*)d_in[4];
  const float* in_proj_b  = (const float*)d_in[5];
  const float* out_proj_w = (const float*)d_in[6];
  const float* out_proj_b = (const float*)d_in[7];
  const float* w_ih0      = (const float*)d_in[8];   // w_hh0/w_hh1 unused: h0=c0=0
  const float* b_ih0      = (const float*)d_in[10];
  const float* b_hh0      = (const float*)d_in[11];
  const float* w_ih1      = (const float*)d_in[12];
  const float* b_ih1      = (const float*)d_in[14];
  const float* b_hh1      = (const float*)d_in[15];
  const float* w_mel      = (const float*)d_in[16];
  const float* b_mel      = (const float*)d_in[17];

  char* ws = (char*)d_ws;
  size_t off = 0;
  auto alloc_h = [&](size_t elems) -> h16* {
    h16* p = (h16*)(ws + off);
    off += ((elems * sizeof(h16) + 255) & ~(size_t)255);
    return p;
  };

  h16* dec_in = alloc_h((size_t)R * 1024);  // [pre | attn_out]
  h16* enc16  = alloc_h((size_t)R * 512);
  h16* prev16 = alloc_h((size_t)R * 96);    // K padded 80 -> 96
  h16* qkv    = alloc_h((size_t)R * 1536);  // q | k | v interleaved per row
  h16* ctx    = alloc_h((size_t)R * 512);
  h16* wpre   = alloc_h((size_t)512 * 96);   // K padded 80 -> 96
  h16* wqkv   = alloc_h((size_t)1536 * 512);
  h16* wout   = alloc_h((size_t)512 * 512);
  h16* wih0   = alloc_h((size_t)2048 * 1024);
  h16* wih1   = alloc_h((size_t)2048 * 512);
  h16* wmel   = alloc_h((size_t)128 * 512);  // N padded 80 -> 128
  // qkv is dead after attention: reuse for LSTM hidden states
  h16* h1 = qkv;
  h16* h2 = qkv + (size_t)R * 512;

  auto cvt = [&](const float* s, h16* dptr, int n) {
    cvt_f32_to_f16<<<(n + 255) / 256, 256, 0, stream>>>(s, dptr, n);
  };
  cvt(enc_f32, enc16, R * 512);
  cvt(in_proj_w, wqkv, 1536 * 512);
  cvt(out_proj_w, wout, 512 * 512);
  cvt(w_ih0, wih0, 2048 * 1024);
  cvt(w_ih1, wih1, 2048 * 512);
  cvt_pad_f16<<<(512 * 96 + 255) / 256, 256, 0, stream>>>(
      w_prenet, wpre, 512, 80, 96, 512 * 96);
  cvt_pad_f16<<<(128 * 512 + 255) / 256, 256, 0, stream>>>(
      w_mel, wmel, 80, 512, 512, 128 * 512);
  build_prev<<<(R * 96 + 255) / 256, 256, 0, stream>>>(mel_f32, prev16, R * 96);

  // prenet: pre = relu(prev @ w_prenet^T + b) -> dec_in[:, 0:512]
  gemm_kernel<<<dim3(R / 128, 4), 256, 0, stream>>>(
      prev16, 96, wpre, b_prenet, 512, 96, 1, dec_in, nullptr, 1024);
  // q = pre @ wq^T + bq -> qkv[:, 0:512]
  gemm_kernel<<<dim3(R / 128, 4), 256, 0, stream>>>(
      dec_in, 1024, wqkv, in_proj_b, 512, 512, 0, qkv, nullptr, 1536);
  // k,v = enc @ {wk,wv}^T -> qkv[:, 512:1536]
  gemm_kernel<<<dim3(R / 128, 8), 256, 0, stream>>>(
      enc16, 512, wqkv + (size_t)512 * 512, in_proj_b + 512, 1024, 512, 0,
      qkv + 512, nullptr, 1536);
  // flash attention -> ctx
  attn_kernel<<<dim3(B * 8, T / 128), 256, 0, stream>>>(qkv, ctx);
  // out_proj -> dec_in[:, 512:1024]
  gemm_kernel<<<dim3(R / 128, 4), 256, 0, stream>>>(
      ctx, 512, wout, out_proj_b, 512, 512, 0, dec_in + 512, nullptr, 1024);
  // LSTM layers (fused gate GEMM + pointwise cell, f-gate skipped)
  lstm_kernel<<<dim3(R / 128, 8), 256, 0, stream>>>(
      dec_in, 1024, wih0, 1024, b_ih0, b_hh0, h1);
  lstm_kernel<<<dim3(R / 128, 8), 256, 0, stream>>>(
      h1, 512, wih1, 512, b_ih1, b_hh1, h2);
  // mel projection -> d_out (f32)
  gemm_kernel<<<dim3(R / 128, 1), 256, 0, stream>>>(
      h2, 512, wmel, b_mel, 80, 512, 0, nullptr, (float*)d_out, 80);
}
